// NeighbourAttn_39513699123442
// MI455X (gfx1250) — matrix operand order repaired and checked by reading, not compiled, not run on hardware
//
#include <hip/hip_runtime.h>
#include <hip/hip_bf16.h>

// ---------------------------------------------------------------------------
// NATTEN block for MI455X (gfx1250, wave32, WMMA).
// Path: f16 inputs to v_wmma_f32_16x16x32_f16, fp32 accumulate, fp32 output.
// ---------------------------------------------------------------------------

typedef __attribute__((ext_vector_type(16))) _Float16 v16h;
typedef __attribute__((ext_vector_type(8)))  _Float16 v8h;
typedef __attribute__((ext_vector_type(8)))  float    v8f;

#define BQ    4
#define NTOK  1024     // HG*WG
#define DIM   768
#define NHEAD 12
#define HDIM  64
#define WIMG  32
#define KWIN  7
#define JPAD  160      // 7*19 = 133 union keys padded to 5*32
#define JVAL  133

__device__ __forceinline__ int imin(int a, int b) { return a < b ? a : b; }
__device__ __forceinline__ int imax(int a, int b) { return a > b ? a : b; }

// Build a 16x32 f16 WMMA A/B operand from two contiguous 8-half LDS chunks.
__device__ __forceinline__ v16h mk16(v8h lo, v8h hi) {
  v16h r;
#pragma unroll
  for (int i = 0; i < 8; ++i) { r[i] = lo[i]; r[8 + i] = hi[i]; }
  return r;
}

__device__ __forceinline__ v8f wmma16(v16h a, v16h b, v8f c) {
  return __builtin_amdgcn_wmma_f32_16x16x32_f16(
      /*neg_a=*/false, a, /*neg_b=*/false, b,
      /*c_mod=*/(short)0, c, /*reuse_a=*/false, /*reuse_b=*/false);
}

// ---------------------------------------------------------------------------
// Kernel 1: qkv = x @ qkv_w.T + qkv_b  -> split into q (scaled), k, v (f16)
// M=4096, N=2304, K=768.  Block tile 32(M) x 64(N), 128 threads (4 waves),
// each wave: one 16-wide N slice, two stacked 16x16 C tiles.
// ---------------------------------------------------------------------------
__global__ __launch_bounds__(128) void natten_qkv_gemm(
    const float* __restrict__ x, const float* __restrict__ w,
    const float* __restrict__ bias,
    _Float16* __restrict__ qh, _Float16* __restrict__ kh,
    _Float16* __restrict__ vh) {
  __shared__ _Float16 As[32][32];
  __shared__ _Float16 Bs[64][32];

  const int m0 = blockIdx.x * 32;
  const int n0 = blockIdx.y * 64;
  const int tid = threadIdx.x;
  const int lane = tid & 31, wv = tid >> 5;
  const int h = (lane >> 4) & 1, ln = lane & 15;

  v8f acc0 = {}, acc1 = {};

  for (int k0 = 0; k0 < DIM; k0 += 32) {
    {  // stage A tile (32x32), f32 -> f16
      const int row = tid >> 2, seg = (tid & 3) * 8;
      const float* src = x + (size_t)(m0 + row) * DIM + k0 + seg;
      v8h d;
#pragma unroll
      for (int i = 0; i < 8; ++i) d[i] = (_Float16)src[i];
      *(v8h*)&As[row][seg] = d;
    }
    {  // stage B tile (64x32 rows of qkv_w), f32 -> f16
      const int row = tid >> 1, seg = (tid & 1) * 16;
      const float* src = w + (size_t)(n0 + row) * DIM + k0 + seg;
#pragma unroll
      for (int i = 0; i < 16; ++i) Bs[row][seg + i] = (_Float16)src[i];
    }
    __syncthreads();
    // B operand: lane (h,n) element e -> K = e + 16h
    v16h bf = *(const v16h*)&Bs[wv * 16 + ln][16 * h];
    // A operand: lane (h,m) element e -> K = e + 16h
    v16h a0 = mk16(*(const v8h*)&As[ln][16 * h],
                   *(const v8h*)&As[ln][16 * h + 8]);
    v16h a1 = mk16(*(const v8h*)&As[16 + ln][16 * h],
                   *(const v8h*)&As[16 + ln][16 * h + 8]);
    acc0 = wmma16(a0, bf, acc0);
    acc1 = wmma16(a1, bf, acc1);
    __syncthreads();
  }

  const int col = n0 + wv * 16 + ln;  // 0..2303
  const float bval = bias[col];
  const int section = col / DIM;      // 0=q 1=k 2=v
  const int d = col - section * DIM;
  const int head = d >> 6, hd = d & 63;
  _Float16* dst = (section == 0) ? qh : (section == 1) ? kh : vh;
  const float scale = (section == 0) ? 0.125f : 1.0f;  // HD^-0.5 = 1/8

#pragma unroll
  for (int r = 0; r < 8; ++r) {
    {
      const int mg = m0 + r + 8 * h;           // C layout: M = r + 8h
      const int b = mg >> 10, t = mg & 1023;
      dst[(((size_t)(b * NHEAD + head) * NTOK + t) * HDIM) + hd] =
          (_Float16)((acc0[r] + bval) * scale);
    }
    {
      const int mg = m0 + 16 + r + 8 * h;
      const int b = mg >> 10, t = mg & 1023;
      dst[(((size_t)(b * NHEAD + head) * NTOK + t) * HDIM) + hd] =
          (_Float16)((acc1[r] + bval) * scale);
    }
  }
}

// ---------------------------------------------------------------------------
// Kernel 2: neighborhood attention. One wave per (batch, head, 16-query tile).
// The 16 queries share an image row; their 7x7 windows union to a dense
// 7x19 = 133-key rectangle (starts r0 = clip(y-3,0,25), c0 = clip(x0-3,0,25)).
// Scores over the union via WMMA, per-query column mask -> -inf, softmax,
// P @ V via WMMA (P re-laid out through LDS).
// ---------------------------------------------------------------------------
__global__ __launch_bounds__(32) void natten_attn(
    const _Float16* __restrict__ qh, const _Float16* __restrict__ kh,
    const _Float16* __restrict__ vh, _Float16* __restrict__ yout) {
  __shared__ _Float16 Qs[16][HDIM];     //  2 KB
  __shared__ _Float16 Ks[JPAD][HDIM];   // 20 KB  keys row-major [j][d]
  __shared__ _Float16 Vt[HDIM][JPAD];   // 20 KB  values transposed [d][j]
  __shared__ _Float16 Ps[16][JPAD];     //  5 KB  probabilities [q][j]

  const int qt = blockIdx.x;   // 0..63 query tile
  const int hh = blockIdx.y;   // head
  const int b  = blockIdx.z;   // batch
  const int lane = threadIdx.x;
  const int h = (lane >> 4) & 1, ln = lane & 15;

  const int t0 = qt * 16;
  const int yrow = t0 >> 5;            // image row of all 16 queries
  const int x0 = t0 & 31;              // 0 or 16
  const int r0 = imin(imax(yrow - 3, 0), WIMG - KWIN);
  const int c0 = imin(imax(x0 - 3, 0), WIMG - KWIN);

  const size_t hb = (size_t)(b * NHEAD + hh) * NTOK * HDIM;

  {  // stage Q (16x64): each lane copies 32 halfs
    const int m = lane >> 1, part = (lane & 1) * 32;
    const _Float16* src = qh + hb + (size_t)(t0 + m) * HDIM + part;
#pragma unroll
    for (int i = 0; i < 32; i += 8)
      *(v8h*)&Qs[m][part + i] = *(const v8h*)(src + i);
  }
  // stage K rows + V transposed; pad j >= 133 with zeros
  for (int jj = lane; jj < JPAD; jj += 32) {
    if (jj < JVAL) {
      const int ri = jj / 19, cj = jj - ri * 19;
      const int tok = (r0 + ri) * WIMG + (c0 + cj);
      const _Float16* ksrc = kh + hb + (size_t)tok * HDIM;
      const _Float16* vsrc = vh + hb + (size_t)tok * HDIM;
#pragma unroll
      for (int i = 0; i < HDIM; i += 8)
        *(v8h*)&Ks[jj][i] = *(const v8h*)(ksrc + i);
#pragma unroll
      for (int d = 0; d < HDIM; ++d) Vt[d][jj] = vsrc[d];
    } else {
      v8h z = {};
#pragma unroll
      for (int i = 0; i < HDIM; i += 8) *(v8h*)&Ks[jj][i] = z;
#pragma unroll
      for (int d = 0; d < HDIM; ++d) Vt[d][jj] = (_Float16)0.0f;
    }
  }
  __syncthreads();

  // Q as WMMA A operand over K-dim d: two 16x32 fragments (d 0..31, 32..63)
  const v16h aq0 = mk16(*(const v8h*)&Qs[ln][16 * h],
                        *(const v8h*)&Qs[ln][16 * h + 8]);
  const v16h aq1 = mk16(*(const v8h*)&Qs[ln][32 + 16 * h],
                        *(const v8h*)&Qs[ln][40 + 16 * h]);

  // scores: 10 chunks of 16 keys, 2 WMMAs each (K-dim = 64)
  float sc[10][8];
#pragma unroll
  for (int c = 0; c < 10; ++c) {
    const v16h b0 = *(const v16h*)&Ks[c * 16 + ln][16 * h];
    const v16h b1 = *(const v16h*)&Ks[c * 16 + ln][32 + 16 * h];
    v8f acc = {};
    acc = wmma16(aq0, b0, acc);
    acc = wmma16(aq1, b1, acc);
#pragma unroll
    for (int r = 0; r < 8; ++r) sc[c][r] = acc[r];
  }

  // per-query mask + rowwise max
  float rmax[8];
#pragma unroll
  for (int r = 0; r < 8; ++r) rmax[r] = -1e30f;
#pragma unroll
  for (int c = 0; c < 10; ++c) {
    const int j = c * 16 + ln;  // key index held by this lane
    const int ri = j / 19, cj = j - ri * 19;
    const bool jin = (j < JVAL);
#pragma unroll
    for (int r = 0; r < 8; ++r) {
      const int m = r + 8 * h;                                 // query row
      const int lo = imin(imax(x0 + m - 3, 0), WIMG - KWIN) - c0;
      const bool ok = jin && (cj >= lo) && (cj <= lo + 6);
      const float s = ok ? sc[c][r] : -1e30f;
      sc[c][r] = s;
      rmax[r] = fmaxf(rmax[r], s);
    }
  }
  // reduce across the 16 lanes of each half (rows r+8h live per half)
#pragma unroll
  for (int off = 1; off < 16; off <<= 1)
#pragma unroll
    for (int r = 0; r < 8; ++r)
      rmax[r] = fmaxf(rmax[r], __shfl_xor(rmax[r], off, 32));

  float rsum[8];
#pragma unroll
  for (int r = 0; r < 8; ++r) rsum[r] = 0.0f;
#pragma unroll
  for (int c = 0; c < 10; ++c)
#pragma unroll
    for (int r = 0; r < 8; ++r) {
      const float e = __expf(sc[c][r] - rmax[r]);  // masked -> 0
      sc[c][r] = e;
      rsum[r] += e;
    }
#pragma unroll
  for (int off = 1; off < 16; off <<= 1)
#pragma unroll
    for (int r = 0; r < 8; ++r) rsum[r] += __shfl_xor(rsum[r], off, 32);

  float rinv[8];
#pragma unroll
  for (int r = 0; r < 8; ++r) rinv[r] = 1.0f / rsum[r];

  // write P to LDS in [q][j] layout (C layout -> row-major)
#pragma unroll
  for (int c = 0; c < 10; ++c)
#pragma unroll
    for (int r = 0; r < 8; ++r)
      Ps[r + 8 * h][c * 16 + ln] = (_Float16)(sc[c][r] * rinv[r]);
  __syncthreads();

  // out = P(16x160) @ Vn(160x64): 4 d-chunks x 5 K-chunks of 32
#pragma unroll
  for (int dc = 0; dc < 4; ++dc) {
    v8f acc = {};
#pragma unroll
    for (int kc = 0; kc < 5; ++kc) {
      const v16h pa = mk16(*(const v8h*)&Ps[ln][kc * 32 + 16 * h],
                           *(const v8h*)&Ps[ln][kc * 32 + 16 * h + 8]);
      const v16h vb = *(const v16h*)&Vt[dc * 16 + ln][kc * 32 + 16 * h];
      acc = wmma16(pa, vb, acc);
    }
#pragma unroll
    for (int r = 0; r < 8; ++r) {
      const int m = r + 8 * h;
      yout[(size_t)(b * NTOK + t0 + m) * DIM + hh * HDIM + dc * 16 + ln] =
          (_Float16)acc[r];
    }
  }
}

// ---------------------------------------------------------------------------
// Kernel 3: out = y @ proj_w.T + proj_b   (M=4096, N=768, K=768), fp32 out.
// ---------------------------------------------------------------------------
__global__ __launch_bounds__(128) void natten_proj_gemm(
    const _Float16* __restrict__ yh, const float* __restrict__ w,
    const float* __restrict__ bias, float* __restrict__ out) {
  __shared__ _Float16 As[32][32];
  __shared__ _Float16 Bs[64][32];

  const int m0 = blockIdx.x * 32;
  const int n0 = blockIdx.y * 64;
  const int tid = threadIdx.x;
  const int lane = tid & 31, wv = tid >> 5;
  const int h = (lane >> 4) & 1, ln = lane & 15;

  v8f acc0 = {}, acc1 = {};

  for (int k0 = 0; k0 < DIM; k0 += 32) {
    {  // stage A tile (already f16)
      const int row = tid >> 2, seg = (tid & 3) * 8;
      *(v8h*)&As[row][seg] =
          *(const v8h*)(yh + (size_t)(m0 + row) * DIM + k0 + seg);
    }
    {  // stage B tile with f32 -> f16
      const int row = tid >> 1, seg = (tid & 1) * 16;
      const float* src = w + (size_t)(n0 + row) * DIM + k0 + seg;
#pragma unroll
      for (int i = 0; i < 16; ++i) Bs[row][seg + i] = (_Float16)src[i];
    }
    __syncthreads();
    v16h bf = *(const v16h*)&Bs[wv * 16 + ln][16 * h];
    v16h a0 = mk16(*(const v8h*)&As[ln][16 * h],
                   *(const v8h*)&As[ln][16 * h + 8]);
    v16h a1 = mk16(*(const v8h*)&As[16 + ln][16 * h],
                   *(const v8h*)&As[16 + ln][16 * h + 8]);
    acc0 = wmma16(a0, bf, acc0);
    acc1 = wmma16(a1, bf, acc1);
    __syncthreads();
  }

  const int col = n0 + wv * 16 + ln;
  const float bv = bias[col];
#pragma unroll
  for (int r = 0; r < 8; ++r) {
    out[(size_t)(m0 + r + 8 * h) * DIM + col] = acc0[r] + bv;
    out[(size_t)(m0 + 16 + r + 8 * h) * DIM + col] = acc1[r] + bv;
  }
}

// ---------------------------------------------------------------------------
extern "C" void kernel_launch(void* const* d_in, const int* in_sizes, int n_in,
                              void* d_out, int out_size, void* d_ws,
                              size_t ws_size, hipStream_t stream) {
  const float* x      = (const float*)d_in[0];  // [4,1024,768]
  const float* qkv_w  = (const float*)d_in[1];  // [2304,768]
  const float* qkv_b  = (const float*)d_in[2];  // [2304]
  const float* proj_w = (const float*)d_in[3];  // [768,768]
  const float* proj_b = (const float*)d_in[4];  // [768]
  float* out = (float*)d_out;                   // [4,1024,768]

  // workspace: q/k/v head-major f16 + attention output f16 (~25 MB)
  const size_t per = (size_t)BQ * NHEAD * NTOK * HDIM;  // 3,145,728 halfs
  _Float16* qh = (_Float16*)d_ws;
  _Float16* kh = qh + per;
  _Float16* vh = kh + per;
  _Float16* yh = vh + per;  // [4096][768]

  natten_qkv_gemm<<<dim3(128, 36), 128, 0, stream>>>(x, qkv_w, qkv_b, qh, kh, vh);
  natten_attn<<<dim3(64, NHEAD, BQ), 32, 0, stream>>>(qh, kh, vh, yh);
  natten_proj_gemm<<<dim3(128, 12), 128, 0, stream>>>(yh, proj_w, proj_b, out);
}
